// Qwen2VisionModel_43885975830842
// MI455X (gfx1250) — matrix operand order repaired and checked by reading, not compiled, not run on hardware
//
#include <hip/hip_runtime.h>
#include <hip/hip_bf16.h>
#include <math.h>

// ---------------------------------------------------------------------------
// Types for CDNA5 WMMA (wave32): V_WMMA_F32_16X16X32_BF16
// ---------------------------------------------------------------------------
typedef __attribute__((ext_vector_type(16))) __bf16 v16bf;
typedef __attribute__((ext_vector_type(8)))  float  v8f;

union AB16 { v16bf v; uint4 q[2]; };

// A-matrix 16x32 bf16 fragment (ISA 7.12.2): lane m = lane&15;
// lanes 0-15 hold K = {0..7, 16..23}, lanes 16-31 hold K = {8..15, 24..31}.
static __device__ __forceinline__ v16bf load_a_frag(const __bf16* row, int lane) {
  const int kb = (lane < 16) ? 0 : 8;
  AB16 f;
  f.q[0] = *(const uint4*)(row + kb);        // K = kb .. kb+7
  f.q[1] = *(const uint4*)(row + kb + 16);   // K = kb+16 .. kb+23
  return f.v;
}
// B-matrix 32x16 bf16 fragment: lane n = lane&15; lanes 0-15 hold K={0..7,16..23},
// lanes 16-31 hold K={8..15,24..31} (same K order as A).
static __device__ __forceinline__ v16bf load_b_frag(const __bf16* row, int lane) {
  const int kb = (lane < 16) ? 0 : 8;
  AB16 f;
  f.q[0] = *(const uint4*)(row + kb);
  f.q[1] = *(const uint4*)(row + kb + 16);
  return f.v;
}

static __device__ __forceinline__ v8f wmma_bf16(v16bf a, v16bf b, v8f c) {
  return __builtin_amdgcn_wmma_f32_16x16x32_bf16(false, a, false, b, (short)0, c,
                                                 false, false);
}

// CDNA5 async global->LDS copy (ASYNCcnt-tracked, no VGPR round trip).
// LDS destination = 32-bit LDS offset (low 32 bits of an LDS flat pointer).
static __device__ __forceinline__ void async_load_b128(const void* g,
                                                       unsigned lds_off) {
  asm volatile("global_load_async_to_lds_b128 %0, %1, off"
               :: "v"(lds_off), "v"((unsigned long long)(size_t)g)
               : "memory");
}
static __device__ __forceinline__ void wait_async0() {
  asm volatile("s_wait_asynccnt 0x0" ::: "memory");
}

// ---------------------------------------------------------------------------
// Model constants (reference's fixed GRID = 2 x (1,32,32))
// ---------------------------------------------------------------------------
#define N_TOK   2048
#define SEG_LEN 1024
#define EMBED   1280
#define HEADS   16
#define HD      80
#define QKV_DIM 3840
#define MLP_DIM 5120
#define OUT_DIM 3584
#define PATCH_IN 1176
#define PATCH_PAD 1184   // 37*32: K padded so async A tiles never read a tail
#define MERGED_TOK 512
#define MERGED_DIM 5120

enum { OP_NONE = 0, OP_RES = 1, OP_QGELU = 2, OP_GELU = 3 };

// ---------------------------------------------------------------------------
// GEMM: C[M,N] = act(A[M,KA(bf16)] @ W[N,K(f32)]^T + bias (+ Res))
// A arrives bf16 (produced by upstream epilogues) and is tile-staged with
// GLOBAL_LOAD_ASYNC_TO_LDS_B128, double-buffered (prefetch k+1 during k).
// W is f32 and staged through VALU v_cvt_pk_bf16_f32 (weights stay f32 in
// HBM: model is compute-bound, converting 247MB of weights per call would
// cost more HBM time than it saves). KA = K rounded up to 32; A cols >= K
// must be zero (NaN would poison WMMA). M%128==0, N%128==0.
// ---------------------------------------------------------------------------
template <int OP, bool OBF>
__global__ __launch_bounds__(256) void gemm_bf16_wmma(
    const __bf16* __restrict__ A, const float* __restrict__ W,
    const float* __restrict__ bias, const float* __restrict__ Res,
    void* __restrict__ Cv, int M, int N, int K, int KA) {
  __shared__ __bf16 As[2][128][32];
  __shared__ __bf16 Ws[2][128][32];

  const int tid  = threadIdx.x;
  const int lane = tid & 31;
  const int wv   = tid >> 5;
  const int wm   = (wv >> 2) * 64;  // wave row offset in block tile
  const int wn   = (wv & 3) * 32;   // wave col offset in block tile
  const long bm  = (long)blockIdx.y * 128;
  const long bn  = (long)blockIdx.x * 128;

  const v8f vzero = {0.f, 0.f, 0.f, 0.f, 0.f, 0.f, 0.f, 0.f};
  v8f acc[4][2];
#pragma unroll
  for (int mi = 0; mi < 4; ++mi)
#pragma unroll
    for (int ni = 0; ni < 2; ++ni) acc[mi][ni] = vzero;

  // wave wv async-copies rows [wv*16, wv*16+16) of the A tile: 2x B128/lane
  auto issueA = [&](int k0, int b) {
#pragma unroll
    for (int t = 0; t < 2; ++t) {
      const int row = wv * 16 + t * 8 + (lane >> 2);
      const int seg = (lane & 3) * 8;  // bf16 elements (16B segments)
      async_load_b128(A + (bm + row) * (long)KA + k0 + seg,
                      (unsigned)(size_t)&As[b][row][seg]);
    }
  };

  issueA(0, 0);
  int buf = 0;
  for (int k0 = 0; k0 < KA; k0 += 32, buf ^= 1) {
    // stage W tile f32 -> bf16 (zero K tail)
    for (int i = tid; i < 128 * 32; i += 256) {
      const int r = i >> 5, c = i & 31;
      const int k = k0 + c;
      Ws[buf][r][c] = (k < K) ? (__bf16)W[(bn + r) * (long)K + k] : (__bf16)0.f;
    }
    wait_async0();       // this wave's A-tile portion landed in LDS
    __syncthreads();     // everyone's A portions + W stores visible
    if (k0 + 32 < KA) issueA(k0 + 32, buf ^ 1);  // prefetch next tile

    v16bf af[4], bfg[2];
#pragma unroll
    for (int mi = 0; mi < 4; ++mi)
      af[mi] = load_a_frag(&As[buf][wm + mi * 16 + (lane & 15)][0], lane);
#pragma unroll
    for (int ni = 0; ni < 2; ++ni)
      bfg[ni] = load_b_frag(&Ws[buf][wn + ni * 16 + (lane & 15)][0], lane);

#pragma unroll
    for (int mi = 0; mi < 4; ++mi)
#pragma unroll
      for (int ni = 0; ni < 2; ++ni)
        acc[mi][ni] = wmma_bf16(af[mi], bfg[ni], acc[mi][ni]);
  }

  // C/D layout: VGPR r -> M = r (lanes 0-15) or r+8 (lanes 16-31); N = lane&15
  const int rbase = (lane >= 16) ? 8 : 0;
  const int cn    = lane & 15;
#pragma unroll
  for (int mi = 0; mi < 4; ++mi)
#pragma unroll
    for (int ni = 0; ni < 2; ++ni)
#pragma unroll
      for (int r = 0; r < 8; ++r) {
        const long row = bm + wm + mi * 16 + rbase + r;
        const long col = bn + wn + ni * 16 + cn;
        float v = acc[mi][ni][r];
        if (bias) v += bias[col];
        if (OP == OP_RES)   v += Res[row * (long)N + col];
        if (OP == OP_QGELU) v = v / (1.f + __expf(-1.702f * v));
        if (OP == OP_GELU)  v = 0.5f * v * (1.f + erff(v * 0.70710678118f));
        if (OBF) ((__bf16*)Cv)[row * (long)N + col] = (__bf16)v;
        else     ((float*)Cv)[row * (long)N + col]  = v;
      }
}

// ---------------------------------------------------------------------------
// LayerNorm (f32 in, bf16 out: feeds GEMM A-side)
// ---------------------------------------------------------------------------
__global__ __launch_bounds__(256) void layernorm_kernel(
    const float* __restrict__ x, const float* __restrict__ w,
    const float* __restrict__ b, __bf16* __restrict__ y, int cols) {
  __shared__ float red[256];
  const int row = blockIdx.x;
  const float* xr = x + (long)row * cols;

  float s = 0.f;
  for (int c = threadIdx.x; c < cols; c += 256) s += xr[c];
  red[threadIdx.x] = s;
  __syncthreads();
  for (int o = 128; o > 0; o >>= 1) {
    if (threadIdx.x < o) red[threadIdx.x] += red[threadIdx.x + o];
    __syncthreads();
  }
  const float mu = red[0] / cols;
  __syncthreads();

  float v = 0.f;
  for (int c = threadIdx.x; c < cols; c += 256) {
    const float d = xr[c] - mu;
    v += d * d;
  }
  red[threadIdx.x] = v;
  __syncthreads();
  for (int o = 128; o > 0; o >>= 1) {
    if (threadIdx.x < o) red[threadIdx.x] += red[threadIdx.x + o];
    __syncthreads();
  }
  const float rstd = rsqrtf(red[0] / cols + 1e-6f);

  for (int c = threadIdx.x; c < cols; c += 256)
    y[(long)row * cols + c] = (__bf16)((xr[c] - mu) * rstd * w[c] + b[c]);
}

// ---------------------------------------------------------------------------
// Pixel f32 -> bf16 with K padded 1176 -> 1184 (zero tail)
// ---------------------------------------------------------------------------
__global__ void cvt_pix_kernel(const float* __restrict__ in,
                               __bf16* __restrict__ out) {
  const int i = blockIdx.x * blockDim.x + threadIdx.x;
  if (i >= N_TOK * PATCH_PAD) return;
  const int row = i / PATCH_PAD, c = i % PATCH_PAD;
  out[i] = (c < PATCH_IN) ? (__bf16)in[(long)row * PATCH_IN + c] : (__bf16)0.f;
}

// ---------------------------------------------------------------------------
// RoPE cos/sin tables (positions from the fixed 2 x (1,32,32) grid,
// spatial-merge 2x2 order)
// ---------------------------------------------------------------------------
__global__ void rope_setup_kernel(float* __restrict__ cosT,
                                  float* __restrict__ sinT) {
  const int n = blockIdx.x;
  const int d = threadIdx.x;
  if (d >= HD) return;
  const int ii   = n & (SEG_LEN - 1);
  const int wm   = ii & 1;
  const int hm   = (ii >> 1) & 1;
  const int blk  = ii >> 2;
  const int wblk = blk & 15;
  const int hblk = blk >> 4;
  const int hp = hblk * 2 + hm, wp = wblk * 2 + wm;

  const int d2 = d % 40;               // emb = concat(rot, rot)
  const int fi = d2 % 20;
  const float pos = (d2 < 20) ? (float)hp : (float)wp;
  const float inv = __powf(10000.f, -(float)fi / 20.f);
  const float a   = pos * inv;
  cosT[n * HD + d] = __cosf(a);
  sinT[n * HD + d] = __sinf(a);
}

// In-place RoPE on q|k pairs (d, d+40) of one (token, head, q|k)
__global__ void rope_apply_kernel(float* __restrict__ qkv,
                                  const float* __restrict__ cosT,
                                  const float* __restrict__ sinT) {
  const int i = blockIdx.x * blockDim.x + threadIdx.x;
  const int total = N_TOK * HEADS * 40 * 2;
  if (i >= total) return;
  const int d = i % 40;
  const int h = (i / 40) % HEADS;
  const int n = (i / (40 * HEADS)) % N_TOK;
  const int s = i / (40 * HEADS * N_TOK);
  float* p = qkv + (long)n * QKV_DIM + s * EMBED + h * HD;
  const float c1 = cosT[n * HD + d],      s1 = sinT[n * HD + d];
  const float c2 = cosT[n * HD + d + 40], s2 = sinT[n * HD + d + 40];
  const float x1 = p[d], x2 = p[d + 40];
  p[d]      = x1 * c1 - x2 * s1;
  p[d + 40] = x2 * c2 + x1 * s2;
}

// ---------------------------------------------------------------------------
// Flash attention (block-diagonal mask == per-segment key walk). Block =
// 4 waves = 64 queries of one head; wave = 16-query tile; key chunk = 32.
// S contraction 80 padded to 96 (3 WMMA k-steps); O d-dim = 5x16 = 80.
// Output bf16 (feeds proj GEMM A-side).
// ---------------------------------------------------------------------------
__global__ __launch_bounds__(128) void attn_kernel(
    const float* __restrict__ qkv,   // [N_TOK][3840], RoPE applied
    __bf16* __restrict__ out) {      // [N_TOK][1280] bf16
  __shared__ __bf16 Kb[32][96];
  __shared__ __bf16 Vb[80][32];
  __shared__ __bf16 Pb[4][16][32];

  const int tid  = threadIdx.x;
  const int lane = tid & 31;
  const int wv   = tid >> 5;
  const int head = blockIdx.y;
  const int q0   = blockIdx.x * 64 + wv * 16;
  const int kseg = (q0 / SEG_LEN) * SEG_LEN;

  const float scale = rsqrtf((float)HD);

  const int mrow = q0 + (lane & 15);
  const int kbA  = (lane < 16) ? 0 : 8;
  v16bf qf[3];
#pragma unroll
  for (int f = 0; f < 3; ++f) {
    AB16 t;
#pragma unroll
    for (int j = 0; j < 16; ++j) {
      const int d = f * 32 + kbA + ((j < 8) ? j : (j + 8));
      const float v =
          (d < HD) ? qkv[(long)mrow * QKV_DIM + head * HD + d] * scale : 0.f;
      t.v[j] = (__bf16)v;
    }
    qf[f] = t.v;
  }

  const v8f vzero = {0.f, 0.f, 0.f, 0.f, 0.f, 0.f, 0.f, 0.f};
  float mrun[8], lrun[8];
  v8f oacc[5];
#pragma unroll
  for (int r = 0; r < 8; ++r) { mrun[r] = -3.0e38f; lrun[r] = 0.f; }
#pragma unroll
  for (int t = 0; t < 5; ++t) oacc[t] = vzero;

  for (int c = 0; c < SEG_LEN; c += 32) {
    __syncthreads();
    for (int i = tid; i < 32 * 96; i += 128) {
      const int kk = i / 96, d = i % 96;
      Kb[kk][d] = (d < HD)
          ? (__bf16)qkv[(long)(kseg + c + kk) * QKV_DIM + EMBED + head * HD + d]
          : (__bf16)0.f;
    }
    for (int i = tid; i < 32 * HD; i += 128) {
      const int kk = i / HD, d = i % HD;
      Vb[d][kk] =
          (__bf16)qkv[(long)(kseg + c + kk) * QKV_DIM + 2 * EMBED + head * HD + d];
    }
    __syncthreads();

    v8f s[2];
#pragma unroll
    for (int nt = 0; nt < 2; ++nt) {
      s[nt] = vzero;
      const int krow = nt * 16 + (lane & 15);
#pragma unroll
      for (int f = 0; f < 3; ++f)
        s[nt] = wmma_bf16(qf[f], load_b_frag(&Kb[krow][f * 32], lane), s[nt]);
    }

    v8f p[2];
    float alpha[8];
#pragma unroll
    for (int r = 0; r < 8; ++r) {
      float v = fmaxf(s[0][r], s[1][r]);
      for (int off = 8; off > 0; off >>= 1) v = fmaxf(v, __shfl_xor(v, off, 16));
      const float mnew = fmaxf(mrun[r], v);
      const float a    = __expf(mrun[r] - mnew);
      const float p0   = __expf(s[0][r] - mnew);
      const float p1   = __expf(s[1][r] - mnew);
      float rs = p0 + p1;
      for (int off = 8; off > 0; off >>= 1) rs += __shfl_xor(rs, off, 16);
      lrun[r] = lrun[r] * a + rs;
      mrun[r] = mnew;
      alpha[r] = a;
      p[0][r] = p0;
      p[1][r] = p1;
    }
#pragma unroll
    for (int t = 0; t < 5; ++t)
#pragma unroll
      for (int r = 0; r < 8; ++r) oacc[t][r] *= alpha[r];

    const int rbase = (lane >= 16) ? 8 : 0;
#pragma unroll
    for (int nt = 0; nt < 2; ++nt)
#pragma unroll
      for (int r = 0; r < 8; ++r)
        Pb[wv][rbase + r][nt * 16 + (lane & 15)] = (__bf16)p[nt][r];
    const v16bf pf = load_a_frag(&Pb[wv][lane & 15][0], lane);

#pragma unroll
    for (int t = 0; t < 5; ++t)
      oacc[t] = wmma_bf16(pf, load_b_frag(&Vb[t * 16 + (lane & 15)][0], lane),
                          oacc[t]);
  }

  const int rbase = (lane >= 16) ? 8 : 0;
#pragma unroll
  for (int t = 0; t < 5; ++t)
#pragma unroll
    for (int r = 0; r < 8; ++r) {
      const int row = q0 + rbase + r;
      const int d   = t * 16 + (lane & 15);
      out[(long)row * EMBED + head * HD + d] = (__bf16)(oacc[t][r] / lrun[r]);
    }
}

// ---------------------------------------------------------------------------
// Launch: full forward pass
// ---------------------------------------------------------------------------
extern "C" void kernel_launch(void* const* d_in, const int* in_sizes, int n_in,
                              void* d_out, int out_size, void* d_ws,
                              size_t ws_size, hipStream_t stream) {
  const float* pix     = (const float*)d_in[0];
  // d_in[1] = grid_thw (constant 2x(1,32,32); baked into kernels)
  const float* patch_w = (const float*)d_in[2];
  const float* ln1_w   = (const float*)d_in[3];
  const float* ln1_b   = (const float*)d_in[4];
  const float* qkv_w   = (const float*)d_in[5];
  const float* qkv_b   = (const float*)d_in[6];
  const float* proj_w  = (const float*)d_in[7];
  const float* proj_b  = (const float*)d_in[8];
  const float* ln2_w   = (const float*)d_in[9];
  const float* ln2_b   = (const float*)d_in[10];
  const float* fc1_w   = (const float*)d_in[11];
  const float* fc1_b   = (const float*)d_in[12];
  const float* fc2_w   = (const float*)d_in[13];
  const float* fc2_b   = (const float*)d_in[14];
  const float* lnq_w   = (const float*)d_in[15];
  const float* lnq_b   = (const float*)d_in[16];
  const float* mfc1_w  = (const float*)d_in[17];
  const float* mfc1_b  = (const float*)d_in[18];
  const float* mfc2_w  = (const float*)d_in[19];
  const float* mfc2_b  = (const float*)d_in[20];
  float* outp = (float*)d_out;

  // Workspace carve (256B-aligned chunks)
  char* w = (char*)d_ws;
  auto carve = [&](size_t bytes) {
    void* p = w;
    w += (bytes + 255) & ~(size_t)255;
    return p;
  };
  float*  x       = (float*)carve(sizeof(float) * N_TOK * EMBED);
  float*  qkv     = (float*)carve(sizeof(float) * N_TOK * QKV_DIM);
  float*  cosT    = (float*)carve(sizeof(float) * N_TOK * HD);
  float*  sinT    = (float*)carve(sizeof(float) * N_TOK * HD);
  __bf16* h_bf    = (__bf16*)carve(sizeof(__bf16) * N_TOK * EMBED);
  __bf16* attn_bf = (__bf16*)carve(sizeof(__bf16) * N_TOK * EMBED);
  __bf16* mlp_bf  = (__bf16*)carve(sizeof(__bf16) * N_TOK * MLP_DIM);
  __bf16* pix_bf  = (__bf16*)carve(sizeof(__bf16) * N_TOK * PATCH_PAD + 256);

  const dim3 blk256(256), blk128(128);

  cvt_pix_kernel<<<(N_TOK * PATCH_PAD + 255) / 256, blk256, 0, stream>>>(pix,
                                                                         pix_bf);
  rope_setup_kernel<<<N_TOK, 80, 0, stream>>>(cosT, sinT);

  // Patch embed: x = pix @ patch_w^T
  gemm_bf16_wmma<OP_NONE, false>
      <<<dim3(EMBED / 128, N_TOK / 128), blk256, 0, stream>>>(
          pix_bf, patch_w, nullptr, nullptr, x, N_TOK, EMBED, PATCH_IN,
          PATCH_PAD);

  for (int i = 0; i < 4; ++i) {
    layernorm_kernel<<<N_TOK, blk256, 0, stream>>>(
        x, ln1_w + i * EMBED, ln1_b + i * EMBED, h_bf, EMBED);
    gemm_bf16_wmma<OP_NONE, false>
        <<<dim3(QKV_DIM / 128, N_TOK / 128), blk256, 0, stream>>>(
            h_bf, qkv_w + (long)i * QKV_DIM * EMBED, qkv_b + (long)i * QKV_DIM,
            nullptr, qkv, N_TOK, QKV_DIM, EMBED, EMBED);
    rope_apply_kernel<<<(N_TOK * HEADS * 40 * 2 + 255) / 256, blk256, 0,
                        stream>>>(qkv, cosT, sinT);
    attn_kernel<<<dim3(N_TOK / 64, HEADS), blk128, 0, stream>>>(qkv, attn_bf);
    gemm_bf16_wmma<OP_RES, false>
        <<<dim3(EMBED / 128, N_TOK / 128), blk256, 0, stream>>>(
            attn_bf, proj_w + (long)i * EMBED * EMBED, proj_b + (long)i * EMBED,
            x, x, N_TOK, EMBED, EMBED, EMBED);
    layernorm_kernel<<<N_TOK, blk256, 0, stream>>>(
        x, ln2_w + i * EMBED, ln2_b + i * EMBED, h_bf, EMBED);
    gemm_bf16_wmma<OP_QGELU, true>
        <<<dim3(MLP_DIM / 128, N_TOK / 128), blk256, 0, stream>>>(
            h_bf, fc1_w + (long)i * MLP_DIM * EMBED, fc1_b + (long)i * MLP_DIM,
            nullptr, mlp_bf, N_TOK, MLP_DIM, EMBED, EMBED);
    gemm_bf16_wmma<OP_RES, false>
        <<<dim3(EMBED / 128, N_TOK / 128), blk256, 0, stream>>>(
            mlp_bf, fc2_w + (long)i * EMBED * MLP_DIM, fc2_b + (long)i * EMBED,
            x, x, N_TOK, EMBED, MLP_DIM, MLP_DIM);
  }

  // Merger: LN -> contiguous reshape [512, 5120] -> gelu MLP -> out
  layernorm_kernel<<<N_TOK, blk256, 0, stream>>>(x, lnq_w, lnq_b, h_bf, EMBED);
  gemm_bf16_wmma<OP_GELU, true>
      <<<dim3(MLP_DIM / 128, MERGED_TOK / 128), blk256, 0, stream>>>(
          h_bf, mfc1_w, mfc1_b, nullptr, mlp_bf, MERGED_TOK, MLP_DIM,
          MERGED_DIM, MERGED_DIM);
  gemm_bf16_wmma<OP_NONE, false>
      <<<dim3(OUT_DIM / 128, MERGED_TOK / 128), blk256, 0, stream>>>(
          mlp_bf, mfc2_w, mfc2_b, nullptr, outp, MERGED_TOK, OUT_DIM,
          MERGED_DIM, MERGED_DIM);
}
